// GAT_L3_63831803953270
// MI455X (gfx1250) — matrix operand order repaired and checked by reading, not compiled, not run on hardware
//
#include <hip/hip_runtime.h>
#include <hip/hip_bf16.h>
#include <math.h>

typedef __attribute__((ext_vector_type(16))) _Float16 v16h;
typedef __attribute__((ext_vector_type(8)))  _Float16 v8h;
typedef __attribute__((ext_vector_type(8)))  float    v8f;

#define NEG_SLOPE 0.2f

static __device__ __forceinline__ float leaky(float v) {
    return v > 0.f ? v : NEG_SLOPE * v;
}

// float atomic max via integer ordering trick (valid for all non-NaN floats)
static __device__ __forceinline__ void atomicMaxF(float* a, float v) {
    if (v >= 0.f) atomicMax((int*)a, __float_as_int(v));
    else          atomicMin((unsigned int*)a, __float_as_uint(v));
}

// ---------------------------------------------------------------- utilities
__global__ void k_fill(float* __restrict__ p, long n, float v) {
    long i = (long)blockIdx.x * blockDim.x + threadIdx.x;
    if (i < n) p[i] = v;
}

__global__ void k_cvt_f16(const float* __restrict__ in, _Float16* __restrict__ out, long n) {
    long i = (long)blockIdx.x * blockDim.x + threadIdx.x;
    if (i < n) out[i] = (_Float16)in[i];
}

// Wt[m*K + k] = (f16) W[k*M + m]   (W is [K,M] row-major; produce B^T rows)
__global__ void k_transpose_f16(const float* __restrict__ W, _Float16* __restrict__ Wt,
                                int K, int M) {
    long i = (long)blockIdx.x * blockDim.x + threadIdx.x;
    if (i >= (long)K * M) return;
    int m = (int)(i / K);
    int k = (int)(i % K);
    Wt[i] = (_Float16)W[(long)k * M + m];
}

// ---------------------------------------------------------------- WMMA GEMM
// C[N,M] = A[N,K] (f16) * B[K,M] (f16, given as Bt[M,K]) , f32 accumulate.
// One wave computes a 16x64 tile. Requires N%16==0, K%32==0, M%64==0.
__global__ __launch_bounds__(32) void k_gemm_wmma(const _Float16* __restrict__ A,
                                                  const _Float16* __restrict__ Bt,
                                                  float* __restrict__ C,
                                                  int K, int M) {
    const int lane = threadIdx.x & 31;
    const int half = lane >> 4;   // 0 | 1
    const int l16  = lane & 15;
    const long rowTile = blockIdx.x;
    const int  colBase = blockIdx.y * 64;
    const long row = rowTile * 16 + l16;

    v8f acc[4] = {};

    for (int k = 0; k < K; k += 32) {
        // A fragment: 16x32 f16, ISA layout:
        //  lanes 0-15 : halves 0..7 = K[k..k+7],   halves 8..15 = K[k+16..k+23]
        //  lanes16-31 : halves 0..7 = K[k+8..k+15],halves 8..15 = K[k+24..k+31]
        v16h a;
        {
            const _Float16* pa = A + row * K + k + half * 8;
            const v8h lo = *(const v8h*)(pa);
            const v8h hi = *(const v8h*)(pa + 16);
#pragma unroll
            for (int i = 0; i < 8; ++i) { a[i] = lo[i]; a[8 + i] = hi[i]; }
        }
#pragma unroll
        for (int j = 0; j < 4; ++j) {
            // B fragment: 32x16 f16; lane n<16 holds column n, K[k..k+7],K[k+16..k+23];
            // lane n+16 holds column n, K[k+8..k+15],K[k+24..k+31] (same order as A).
            const long col = colBase + j * 16 + l16;
            const _Float16* pb = Bt + col * K + k + half * 8;
            const v8h b0 = *(const v8h*)(pb);
            const v8h b1 = *(const v8h*)(pb + 16);
            v16h b;
#pragma unroll
            for (int i = 0; i < 8; ++i) { b[i] = b0[i]; b[8 + i] = b1[i]; }
            acc[j] = __builtin_amdgcn_wmma_f32_16x16x32_f16(
                false, a, false, b, (short)0, acc[j], false, false);
        }
    }
    // C/D layout: VGPR r -> row (r + half*8), col = l16
#pragma unroll
    for (int j = 0; j < 4; ++j) {
#pragma unroll
        for (int r = 0; r < 8; ++r) {
            const long orow = rowTile * 16 + half * 8 + r;
            C[orow * (long)M + colBase + j * 16 + l16] = acc[j][r];
        }
    }
}

// ---------------------------------------------------------------- attention
// alpha_src[n,h] = <h[n,h,:], a_src[h,:]> ; alpha_dst likewise
__global__ void k_alpha(const float* __restrict__ h,
                        const float* __restrict__ a_src, const float* __restrict__ a_dst,
                        float* __restrict__ asrc, float* __restrict__ adst,
                        int N, int H, int C) {
    int t = blockIdx.x * blockDim.x + threadIdx.x;
    if (t >= N * H) return;
    int n = t / H, hd = t % H;
    const float* hp = h + (long)n * H * C + (long)hd * C;
    const float* as = a_src + (long)hd * C;
    const float* ad = a_dst + (long)hd * C;
    float ss = 0.f, sd = 0.f;
    for (int c = 0; c < C; ++c) { float v = hp[c]; ss += v * as[c]; sd += v * ad[c]; }
    asrc[t] = ss; adst[t] = sd;
}

__global__ void k_edge_max(const int* __restrict__ ei, int E, int N, int H,
                           const float* __restrict__ asrc, const float* __restrict__ adst,
                           float* __restrict__ smax) {
    int e = blockIdx.x * blockDim.x + threadIdx.x;
    if (e >= E + N) return;
    int s, d;
    if (e < E) { s = ei[e]; d = ei[E + e]; } else { s = d = e - E; }
    for (int hd = 0; hd < H; ++hd) {
        float v = leaky(asrc[s * H + hd] + adst[d * H + hd]);
        atomicMaxF(&smax[d * H + hd], v);
    }
}

__global__ void k_edge_expsum(const int* __restrict__ ei, int E, int N, int H,
                              const float* __restrict__ asrc, const float* __restrict__ adst,
                              const float* __restrict__ smax, float* __restrict__ ssum) {
    int e = blockIdx.x * blockDim.x + threadIdx.x;
    if (e >= E + N) return;
    int s, d;
    if (e < E) { s = ei[e]; d = ei[E + e]; } else { s = d = e - E; }
    for (int hd = 0; hd < H; ++hd) {
        float v = leaky(asrc[s * H + hd] + adst[d * H + hd]);
        atomicAdd(&ssum[d * H + hd], __expf(v - smax[d * H + hd]));
    }
}

// one wave per edge: out[dst, :] += alpha[h(c)] * h[src, :]
template <int H, int C>
__global__ __launch_bounds__(256) void k_edge_agg(const int* __restrict__ ei, int E, int N,
                                                  const float* __restrict__ asrc,
                                                  const float* __restrict__ adst,
                                                  const float* __restrict__ smax,
                                                  const float* __restrict__ ssum,
                                                  const float* __restrict__ h,
                                                  float* __restrict__ out) {
    const int lane = threadIdx.x & 31;
    const int e = blockIdx.x * (blockDim.x >> 5) + (threadIdx.x >> 5);
    if (e >= E + N) return;
    int s, d;
    if (e < E) { s = ei[e]; d = ei[E + e]; } else { s = d = e - E; }
    float alpha[H];
#pragma unroll
    for (int hd = 0; hd < H; ++hd) {
        float v = leaky(asrc[s * H + hd] + adst[d * H + hd]);
        alpha[hd] = __expf(v - smax[d * H + hd]) / ssum[d * H + hd];
    }
    const long HC = (long)H * C;
#pragma unroll
    for (int k = 0; k < (H * C) / 32; ++k) {
        const int c = lane + k * 32;
        const int hd = (k * 32) / C;  // valid: C%32==0, lane<32 never crosses a head
        atomicAdd(&out[(long)d * HC + c], alpha[hd] * h[(long)s * HC + c]);
    }
}

__global__ void k_bias_elu_cvt(const float* __restrict__ agg, const float* __restrict__ bias,
                               _Float16* __restrict__ act, long n, int HC) {
    long i = (long)blockIdx.x * blockDim.x + threadIdx.x;
    if (i >= n) return;
    float v = agg[i] + bias[i % HC];
    v = v > 0.f ? v : (__expf(v) - 1.f);  // ELU
    act[i] = (_Float16)v;
}

__global__ void k_add_bias(float* __restrict__ out, const float* __restrict__ b,
                           long n, int C) {
    long i = (long)blockIdx.x * blockDim.x + threadIdx.x;
    if (i >= n) return;
    out[i] = out[i] + b[i % C];
}

// ---------------------------------------------------------------- driver
static inline int cdiv(long a, long b) { return (int)((a + b - 1) / b); }

extern "C" void kernel_launch(void* const* d_in, const int* in_sizes, int n_in,
                              void* d_out, int out_size, void* d_ws, size_t ws_size,
                              hipStream_t stream) {
    const float* x   = (const float*)d_in[0];
    const int*   ei  = (const int*)d_in[1];
    const float* W1  = (const float*)d_in[2];
    const float* as1 = (const float*)d_in[3];
    const float* ad1 = (const float*)d_in[4];
    const float* b1  = (const float*)d_in[5];
    const float* W2  = (const float*)d_in[6];
    const float* as2 = (const float*)d_in[7];
    const float* ad2 = (const float*)d_in[8];
    const float* b2  = (const float*)d_in[9];
    const float* W3  = (const float*)d_in[10];
    const float* as3 = (const float*)d_in[11];
    const float* ad3 = (const float*)d_in[12];
    const float* b3  = (const float*)d_in[13];
    float* out = (float*)d_out;

    const int F_in = 128, H = 4, C = 64, HC = 256;
    const int N = in_sizes[0] / F_in;   // 50000 (divisible by 16)
    const int E = in_sizes[1] / 2;      // 800000
    const int nE = E + N;               // edges incl. self-loops
    const int rowsT = N / 16;

    char* ws = (char*)d_ws;
    float*    bufA = (float*)ws;        ws += (size_t)N * HC * 4;   // GEMM output h
    float*    bufB = (float*)ws;        ws += (size_t)N * HC * 4;   // aggregation output
    _Float16* act  = (_Float16*)ws;     ws += (size_t)N * HC * 2;   // f16 activations
    _Float16* wt   = (_Float16*)ws;     ws += (size_t)HC * HC * 2;  // f16 W^T
    float*    asrc = (float*)ws;        ws += (size_t)N * H * 4;
    float*    adst = (float*)ws;        ws += (size_t)N * H * 4;
    float*    smax = (float*)ws;        ws += (size_t)N * H * 4;
    float*    ssum = (float*)ws;        ws += (size_t)N * H * 4;

    const float NEG_INF = -__builtin_inff();

    // ================= layer 1 (128 -> 4x64, concat) =================
    k_cvt_f16<<<cdiv((long)N * F_in, 256), 256, 0, stream>>>(x, act, (long)N * F_in);
    k_transpose_f16<<<cdiv((long)F_in * HC, 256), 256, 0, stream>>>(W1, wt, F_in, HC);
    k_gemm_wmma<<<dim3(rowsT, HC / 64), 32, 0, stream>>>(act, wt, bufA, F_in, HC);
    k_alpha<<<cdiv((long)N * H, 256), 256, 0, stream>>>(bufA, as1, ad1, asrc, adst, N, H, C);
    k_fill<<<cdiv((long)N * H, 256), 256, 0, stream>>>(smax, (long)N * H, NEG_INF);
    k_fill<<<cdiv((long)N * H, 256), 256, 0, stream>>>(ssum, (long)N * H, 0.f);
    k_fill<<<cdiv((long)N * HC, 256), 256, 0, stream>>>(bufB, (long)N * HC, 0.f);
    k_edge_max<<<cdiv(nE, 256), 256, 0, stream>>>(ei, E, N, H, asrc, adst, smax);
    k_edge_expsum<<<cdiv(nE, 256), 256, 0, stream>>>(ei, E, N, H, asrc, adst, smax, ssum);
    k_edge_agg<4, 64><<<cdiv(nE, 8), 256, 0, stream>>>(ei, E, N, asrc, adst, smax, ssum, bufA, bufB);
    k_bias_elu_cvt<<<cdiv((long)N * HC, 256), 256, 0, stream>>>(bufB, b1, act, (long)N * HC, HC);

    // ================= layer 2 (256 -> 4x64, concat) =================
    k_transpose_f16<<<cdiv((long)HC * HC, 256), 256, 0, stream>>>(W2, wt, HC, HC);
    k_gemm_wmma<<<dim3(rowsT, HC / 64), 32, 0, stream>>>(act, wt, bufA, HC, HC);
    k_alpha<<<cdiv((long)N * H, 256), 256, 0, stream>>>(bufA, as2, ad2, asrc, adst, N, H, C);
    k_fill<<<cdiv((long)N * H, 256), 256, 0, stream>>>(smax, (long)N * H, NEG_INF);
    k_fill<<<cdiv((long)N * H, 256), 256, 0, stream>>>(ssum, (long)N * H, 0.f);
    k_fill<<<cdiv((long)N * HC, 256), 256, 0, stream>>>(bufB, (long)N * HC, 0.f);
    k_edge_max<<<cdiv(nE, 256), 256, 0, stream>>>(ei, E, N, H, asrc, adst, smax);
    k_edge_expsum<<<cdiv(nE, 256), 256, 0, stream>>>(ei, E, N, H, asrc, adst, smax, ssum);
    k_edge_agg<4, 64><<<cdiv(nE, 8), 256, 0, stream>>>(ei, E, N, asrc, adst, smax, ssum, bufA, bufB);
    k_bias_elu_cvt<<<cdiv((long)N * HC, 256), 256, 0, stream>>>(bufB, b2, act, (long)N * HC, HC);

    // ================= layer 3 (256 -> 1x64, mean==identity) =========
    k_transpose_f16<<<cdiv((long)HC * C, 256), 256, 0, stream>>>(W3, wt, HC, C);
    k_gemm_wmma<<<dim3(rowsT, 1), 32, 0, stream>>>(act, wt, bufA, HC, C);   // M = 64
    k_alpha<<<cdiv((long)N, 256), 256, 0, stream>>>(bufA, as3, ad3, asrc, adst, N, 1, C);
    k_fill<<<cdiv((long)N, 256), 256, 0, stream>>>(smax, (long)N, NEG_INF);
    k_fill<<<cdiv((long)N, 256), 256, 0, stream>>>(ssum, (long)N, 0.f);
    k_fill<<<cdiv((long)N * C, 256), 256, 0, stream>>>(out, (long)N * C, 0.f);
    k_edge_max<<<cdiv(nE, 256), 256, 0, stream>>>(ei, E, N, 1, asrc, adst, smax);
    k_edge_expsum<<<cdiv(nE, 256), 256, 0, stream>>>(ei, E, N, 1, asrc, adst, smax, ssum);
    k_edge_agg<1, 64><<<cdiv(nE, 8), 256, 0, stream>>>(ei, E, N, asrc, adst, smax, ssum, bufA, out);
    k_add_bias<<<cdiv((long)N * C, 256), 256, 0, stream>>>(out, b3, (long)N * C, C);
}
